// AlignConv_16192026706266
// MI455X (gfx1250) — matrix operand order repaired and checked by reading, not compiled, not run on hardware
//
#include <hip/hip_runtime.h>
#include <hip/hip_bf16.h>

typedef __attribute__((ext_vector_type(16))) __bf16 v16bf;
typedef __attribute__((ext_vector_type(8)))  float  v8f;

union ABu { uint4 q[2]; v16bf v; };

__device__ __forceinline__ unsigned short bf16_rne(float f) {
  unsigned u = __float_as_uint(f);
  unsigned r = u + 0x7FFFu + ((u >> 16) & 1u);
  return (unsigned short)(r >> 16);
}
__device__ __forceinline__ float bf16f(unsigned short h) {
  return __uint_as_float(((unsigned)h) << 16);
}

#define CIN   256
#define COUT  256
#define HH    128
#define WW    128
#define HWSZ  (HH * WW)
#define K2T   9
#define NCHK  8          // 256 channels / 32 per WMMA K
#define PACK_WORDS (K2T * NCHK * 16 * 32 * 8)   // per hi or lo array (u32)

// ---------------------------------------------------------------------------
// Pack weights [Cout, C, 9] (fp32) into WMMA B-fragment layout, bf16 hi/lo.
// Layout: [tap][cchunk][ntile][lane][v] as u32 (pair of bf16: K=16h+2v, +1).
// ---------------------------------------------------------------------------
__global__ void AlignConv_pack_w(const float* __restrict__ w,
                                 unsigned* __restrict__ bhi,
                                 unsigned* __restrict__ blo) {
  int t = blockIdx.x * blockDim.x + threadIdx.x;
  if (t >= K2T * NCHK * 16 * 32) return;
  int L   = t & 31;
  int nt  = (t >> 5) & 15;
  int cc  = (t >> 9) & 7;
  int tap = t >> 12;
  int n = L & 15, h = L >> 4;
  int o = nt * 16 + n;
  unsigned base = (unsigned)t * 8u;
#pragma unroll
  for (int v = 0; v < 8; ++v) {
    int kbase = 16 * h + 2 * v;       // B 32x16: lanes0-15 K=0..15, lanes16-31 K=16..31
    int c0 = cc * 32 + kbase;
    float w0 = w[(o * CIN + c0) * K2T + tap];
    float w1 = w[(o * CIN + c0 + 1) * K2T + tap];
    unsigned short h0 = bf16_rne(w0), h1 = bf16_rne(w1);
    unsigned short l0 = bf16_rne(w0 - bf16f(h0));
    unsigned short l1 = bf16_rne(w1 - bf16f(h1));
    bhi[base + v] = (unsigned)h0 | ((unsigned)h1 << 16);
    blo[base + v] = (unsigned)l0 | ((unsigned)l1 << 16);
  }
}

// ---------------------------------------------------------------------------
// Transpose x [B,C,H,W] -> xT [B,H*W,C] so bilinear gathers are coalesced
// across channels (adjacent lanes = adjacent channels at one sample point).
// ---------------------------------------------------------------------------
__global__ __launch_bounds__(256)
void AlignConv_transpose(const float* __restrict__ x, float* __restrict__ xT) {
  __shared__ float tile[32][33];
  const int tx = threadIdx.x & 31;
  const int ty = threadIdx.x >> 5;          // 0..7
  const int p0 = blockIdx.x * 32;           // pixel tile
  const int c0 = blockIdx.y * 32;           // channel tile
  const int b  = blockIdx.z;
#pragma unroll
  for (int r = 0; r < 4; ++r) {
    int c = c0 + ty + 8 * r;
    tile[ty + 8 * r][tx] = x[((size_t)(b * CIN + c)) * HWSZ + p0 + tx];
  }
  __syncthreads();
#pragma unroll
  for (int r = 0; r < 4; ++r) {
    int p = p0 + ty + 8 * r;
    xT[((size_t)b * HWSZ + p) * CIN + c0 + tx] = tile[tx][ty + 8 * r];
  }
}

// ---------------------------------------------------------------------------
// Fused deformable conv: one block = 16 consecutive pixels x 256 outputs.
// ---------------------------------------------------------------------------
__global__ __launch_bounds__(256)
void AlignConv_main(const float* __restrict__ xT,
                    const float* __restrict__ anchors,
                    const unsigned* __restrict__ Bhi,
                    const unsigned* __restrict__ Blo,
                    float* __restrict__ out) {
  __shared__ __align__(16) unsigned short sAhi[16 * 40];
  __shared__ __align__(16) unsigned short sAlo[16 * 40];
  __shared__ __align__(16) unsigned sIdx[K2T][16][4];   // (pixel*C) element offset
  __shared__ __align__(16) float    sWt [K2T][16][4];
  __shared__ __align__(16) float    sOut[8][16 * 17];

  const int tid  = threadIdx.x;
  const int lane = tid & 31;
  const int wave = tid >> 5;

  const int p0 = blockIdx.x * 16;          // 16 consecutive pixels of one row
  const int b  = p0 >> 14;                 // / (H*W)
  const int i  = (p0 & (HWSZ - 1)) >> 7;   // / W
  const int j0 = p0 & (WW - 1);

  // ---- prologue: bilinear corner offsets + weights per (pixel, tap) ----
  if (tid < 16 * K2T) {
    int m = tid & 15, tap = tid >> 4;
    const float* a = anchors + (size_t)(p0 + m) * 5;
    float xc = a[0] * 0.125f, yc = a[1] * 0.125f;
    float dw = a[2] * (0.125f / 3.0f), dh = a[3] * (0.125f / 3.0f);
    float cs = cosf(a[4]), sn = sinf(a[4]);
    float kx = (float)(tap % 3 - 1), ky = (float)(tap / 3 - 1);
    float px = xc + cs * dw * kx - sn * dh * ky;
    float py = yc + sn * dw * kx + cs * dh * ky;
    float x0 = floorf(px), y0 = floorf(py);
    float wx = px - x0, wy = py - y0;
#pragma unroll
    for (int c4 = 0; c4 < 4; ++c4) {
      int dy = c4 >> 1, dx = c4 & 1;
      float yf = y0 + (float)dy, xf = x0 + (float)dx;
      bool valid = (yf >= 0.f) & (yf <= (float)(HH - 1)) &
                   (xf >= 0.f) & (xf <= (float)(WW - 1));
      float ycl = fminf(fmaxf(yf, 0.f), (float)(HH - 1));
      float xcl = fminf(fmaxf(xf, 0.f), (float)(WW - 1));
      sIdx[tap][m][c4] = (unsigned)(((int)ycl * WW + (int)xcl) << 8);
      float wc = (dy ? wy : 1.f - wy) * (dx ? wx : 1.f - wx);
      sWt[tap][m][c4] = valid ? wc : 0.f;
    }
  }

  v8f acc0 = {}; v8f acc1 = {};
  const int cl = tid & 31;          // channel sub-lane when sampling (coalesced)
  const int mh = tid >> 5;          // wave-uniform pixel: m = mh, mh+8
  const int nt0 = wave * 2;
  const int mm = lane & 15, hh = lane >> 4;
  // wave-uniform base -> SADDR-form gathers with 32-bit VGPR offsets
  const float* xb_base = xT + (size_t)b * HWSZ * CIN;

  for (int tap = 0; tap < K2T; ++tap) {
    for (int cc = 0; cc < NCHK; ++cc) {
      __syncthreads();   // protect sA from previous iteration's readers
      // ---- bilinear sampling: wave-uniform pixel, lanes over 32 channels ----
      const unsigned coff = (unsigned)(cc * 32 + cl);
#pragma unroll
      for (int mp = 0; mp < 2; ++mp) {
        int m = mh + 8 * mp;
        uint4  id = *(const uint4*) sIdx[tap][m];   // wave-uniform -> coalesced
        float4 wt = *(const float4*)sWt [tap][m];
        float v = wt.x * xb_base[id.x + coff] + wt.y * xb_base[id.y + coff] +
                  wt.z * xb_base[id.z + coff] + wt.w * xb_base[id.w + coff];
        unsigned short hv = bf16_rne(v);
        sAhi[m * 40 + cl] = hv;
        sAlo[m * 40 + cl] = bf16_rne(v - bf16f(hv));
      }
      __syncthreads();
      // ---- A fragments (contiguous k order used here: K=16h+2v) ----
      const unsigned short* rh = &sAhi[mm * 40 + 16 * hh];
      const unsigned short* rl = &sAlo[mm * 40 + 16 * hh];
      ABu ua, ub;
      ua.q[0] = *(const uint4*)(rh);        ua.q[1] = *(const uint4*)(rh + 8);
      ub.q[0] = *(const uint4*)(rl);        ub.q[1] = *(const uint4*)(rl + 8);
      v16bf Ah = ua.v, Al = ub.v;
      // ---- B fragments (pre-packed, contiguous 32B per lane) ----
      size_t boff = (((size_t)((tap * NCHK + cc) * 16 + nt0) * 32 + lane) << 3);
      const uint4* bph = (const uint4*)(Bhi + boff);
      const uint4* bpl = (const uint4*)(Blo + boff);
      ABu h0, l0, h1, l1;
      h0.q[0] = bph[0];   h0.q[1] = bph[1];
      l0.q[0] = bpl[0];   l0.q[1] = bpl[1];
      h1.q[0] = bph[64];  h1.q[1] = bph[65];   // nt0+1: +256 u32 = +64 uint4
      l1.q[0] = bpl[64];  l1.q[1] = bpl[65];
      // ---- hi/lo split GEMM: 3 WMMAs per N-tile ----
      acc0 = __builtin_amdgcn_wmma_f32_16x16x32_bf16(false, Ah, false, h0.v,
                                                     (short)0, acc0, false, false);
      acc1 = __builtin_amdgcn_wmma_f32_16x16x32_bf16(false, Ah, false, h1.v,
                                                     (short)0, acc1, false, false);
      acc0 = __builtin_amdgcn_wmma_f32_16x16x32_bf16(false, Ah, false, l0.v,
                                                     (short)0, acc0, false, false);
      acc1 = __builtin_amdgcn_wmma_f32_16x16x32_bf16(false, Ah, false, l1.v,
                                                     (short)0, acc1, false, false);
      acc0 = __builtin_amdgcn_wmma_f32_16x16x32_bf16(false, Al, false, h0.v,
                                                     (short)0, acc0, false, false);
      acc1 = __builtin_amdgcn_wmma_f32_16x16x32_bf16(false, Al, false, h1.v,
                                                     (short)0, acc1, false, false);
    }
  }

  // ---- epilogue: ReLU + LDS transpose -> coalesced stores ----
  float* myT = &sOut[wave][0];
  // uniform base + 32-bit offsets for stores
  float* out_base = out + (size_t)b * COUT * HWSZ + (unsigned)(i * WW + j0);
#pragma unroll
  for (int tile = 0; tile < 2; ++tile) {
    v8f a = tile ? acc1 : acc0;
    __syncthreads();
#pragma unroll
    for (int r = 0; r < 8; ++r)                 // D: m = r + 8h, n = lane%16
      myT[(r + 8 * hh) * 17 + mm] = fmaxf(a[r], 0.f);
    __syncthreads();
    int nt = nt0 + tile;
#pragma unroll
    for (int pass = 0; pass < 8; ++pass) {
      int ol = pass * 2 + hh;                   // output-channel within tile
      float vv = myT[mm * 17 + ol];             // pixel j = j0+mm, channel nt*16+ol
      out_base[(unsigned)((nt * 16 + ol) * HWSZ + mm)] = vv;
    }
  }
}

extern "C" void kernel_launch(void* const* d_in, const int* in_sizes, int n_in,
                              void* d_out, int out_size, void* d_ws, size_t ws_size,
                              hipStream_t stream) {
  const float* x       = (const float*)d_in[0];
  const float* anchors = (const float*)d_in[1];
  const float* weight  = (const float*)d_in[2];
  float* out = (float*)d_out;

  // ws layout: [ Bhi | Blo | xT ]  = 1.18MB + 1.18MB + 33.5MB
  unsigned* bhi = (unsigned*)d_ws;
  unsigned* blo = bhi + PACK_WORDS;
  float*    xT  = (float*)(blo + PACK_WORDS);

  // 1) pack weights into WMMA B-fragment layout (bf16 hi/lo)
  {
    int total = K2T * NCHK * 16 * 32;
    AlignConv_pack_w<<<(total + 255) / 256, 256, 0, stream>>>(weight, bhi, blo);
  }
  // 2) transpose x to channels-last for coalesced bilinear gathers
  {
    dim3 grid(HWSZ / 32, CIN / 32, 2);
    AlignConv_transpose<<<grid, 256, 0, stream>>>(x, xT);
  }
  // 3) fused deformable conv + GEMM + ReLU
  {
    int blocks = (2 * HWSZ) / 16;   // B*H*W / 16 = 2048
    AlignConv_main<<<blocks, 256, 0, stream>>>(xT, anchors, bhi, blo, out);
  }
}
